// MultiHeadAttention_8735963480133
// MI455X (gfx1250) — matrix operand order repaired and checked by reading, not compiled, not run on hardware
//
#include <hip/hip_runtime.h>
#include <hip/hip_bf16.h>

// ---------------------------------------------------------------------------
// MHA forward for MI455X (gfx1250, wave32, WMMA bf16 16x16x32, fp32 accum)
// x:(4,2048,1024) W_qkv:(1024,3072) W_out:(1024,1024)  h=16 d=64
// K-tiles staged by the Tensor Data Mover (tensor_load_to_lds + TENSORcnt).
// ---------------------------------------------------------------------------

typedef __bf16 v16bf __attribute__((ext_vector_type(16)));
typedef __bf16 v8bf  __attribute__((ext_vector_type(8)));
typedef float  v8f   __attribute__((ext_vector_type(8)));
typedef unsigned int u32x4 __attribute__((ext_vector_type(4)));
typedef int          i32x8 __attribute__((ext_vector_type(8)));
typedef int          i32x4 __attribute__((ext_vector_type(4)));

#define DEVINL __device__ __forceinline__

static constexpr int Bsz = 4;
static constexpr int T   = 2048;
static constexpr int E   = 1024;
static constexpr int H   = 16;
static constexpr int D   = 64;
static constexpr int T3  = 3 * E;          // 3072
static constexpr int MROWS = Bsz * T;      // 8192

DEVINL v8f wmma_bf16(v16bf a, v16bf b, v8f c) {
  // v_wmma_f32_16x16x32_bf16: D = A(16x32) * B(32x16) + C(16x16 f32)
  return __builtin_amdgcn_wmma_f32_16x16x32_bf16(
      /*neg_a=*/false, a, /*neg_b=*/false, b,
      /*c_mod=*/(short)0, c, /*reuse_a=*/false, /*reuse_b=*/false);
}

// ds_swizzle_b32, group-of-32 mode: new_lane = (lane & 0x1f) ^ XM
template <int XM>
DEVINL float swz_xor(float v) {
  int i = __float_as_int(v);
  i = __builtin_amdgcn_ds_swizzle(i, 0x1F | (XM << 10));
  return __int_as_float(i);
}

// A fragment (16x32, row-major source with leading dim `ld` in elements).
// ISA layout: lane holds row = lane&15; half = lane>>4 selects K base;
// elements 0..7 -> K = kofs + half*8 + i ; elements 8..15 -> +16.
DEVINL v16bf load_a_frag(const __bf16* base, int ld, int lane, int kofs) {
  int row = lane & 15, half = (lane >> 4) & 1;
  const __bf16* p = base + row * ld + kofs + half * 8;
  v8bf lo = *(const v8bf*)(p);
  v8bf hi = *(const v8bf*)(p + 16);
  v16bf a;
#pragma unroll
  for (int i = 0; i < 8; ++i) { a[i] = lo[i]; a[i + 8] = hi[i]; }
  return a;
}

// B fragment (32x16) read from an LDS tile stored TRANSPOSED: Bt[n][k].
// ISA layout: lane holds col = lane&15; elements 0..7 -> K = kofs + half*8 + i ; elements 8..15 -> +16.
DEVINL v16bf load_b_frag(const __bf16* base, int ld, int lane, int kofs) {
  int col = lane & 15, half = (lane >> 4) & 1;
  const __bf16* p = base + col * ld + kofs + half * 8;
  v8bf b0 = *(const v8bf*)(p);
  v8bf b1 = *(const v8bf*)(p + 16);
  v16bf b;
#pragma unroll
  for (int i = 0; i < 8; ++i) { b[i] = b0[i]; b[i + 8] = b1[i]; }
  return b;
}

// ---------------------------------------------------------------------------
// Generic fp32-in / bf16-compute GEMM:  C(MxN) = A(MxK) * B(KxN)
// Block = 256 threads (8 waves), tile 128x128, K staged 64 at a time in LDS
// (16 WMMAs per barrier pair). Wave (wr,wc) computes a 32x64 sub-tile.
// ---------------------------------------------------------------------------
template <bool OUT_BF16>
__global__ __launch_bounds__(256)
void gemm_wmma_bf16(const float* __restrict__ A, const float* __restrict__ B,
                    void* __restrict__ Cout, int M, int N, int K) {
  constexpr int KT  = 64;                      // K-tile
  constexpr int LDT = 72;                      // 64 + 8 pad (keeps 16B align)
  __shared__ __attribute__((aligned(16))) __bf16 Als[128 * LDT];
  __shared__ __attribute__((aligned(16))) __bf16 Bls[128 * LDT];

  const int tid  = threadIdx.x;
  const int lane = tid & 31;
  const int wave = tid >> 5;
  const int wr   = wave >> 1;                  // 0..3
  const int wc   = wave & 1;                   // 0..1
  const int m0   = blockIdx.x * 128;
  const int n0   = blockIdx.y * 128;

  v8f acc[2][4];
#pragma unroll
  for (int mi = 0; mi < 2; ++mi)
#pragma unroll
    for (int ni = 0; ni < 4; ++ni)
#pragma unroll
      for (int r = 0; r < 8; ++r) acc[mi][ni][r] = 0.0f;

  const int arow = tid >> 1;                   // 0..127
  const int acol = (tid & 1) * 32;             // 0 / 32
  const int bk   = tid & 63;                   // 0..63
  const int bng  = (tid >> 6) * 16;            // 0..48

  for (int kt = 0; kt < K; kt += KT) {
    // ---- stage A tile (128x64) as bf16, row-major ----
    const float* ap = A + (size_t)(m0 + arow) * K + kt + acol;
#pragma unroll
    for (int i = 0; i < 32; i += 4) {
      float4 f = *(const float4*)(ap + i);
      __bf16* d = &Als[arow * LDT + acol + i];
      d[0] = (__bf16)f.x; d[1] = (__bf16)f.y;
      d[2] = (__bf16)f.z; d[3] = (__bf16)f.w;
    }
    // ---- stage B tile (64x128) TRANSPOSED: Bls[n][k] ----
#pragma unroll
    for (int g = 0; g < 2; ++g) {
      const int bn = bng + g * 64;             // two 16-col groups per thread
      const float* bp = B + (size_t)(kt + bk) * N + n0 + bn;
#pragma unroll
      for (int i = 0; i < 16; i += 4) {
        float4 f = *(const float4*)(bp + i);
        Bls[(bn + i + 0) * LDT + bk] = (__bf16)f.x;
        Bls[(bn + i + 1) * LDT + bk] = (__bf16)f.y;
        Bls[(bn + i + 2) * LDT + bk] = (__bf16)f.z;
        Bls[(bn + i + 3) * LDT + bk] = (__bf16)f.w;
      }
    }
    __syncthreads();

#pragma unroll
    for (int kk = 0; kk < KT; kk += 32) {
      v16bf bf[4];
#pragma unroll
      for (int ni = 0; ni < 4; ++ni)
        bf[ni] = load_b_frag(&Bls[(wc * 64 + ni * 16) * LDT], LDT, lane, kk);
#pragma unroll
      for (int mi = 0; mi < 2; ++mi) {
        v16bf af = load_a_frag(&Als[(wr * 32 + mi * 16) * LDT], LDT, lane, kk);
#pragma unroll
        for (int ni = 0; ni < 4; ++ni)
          acc[mi][ni] = wmma_bf16(af, bf[ni], acc[mi][ni]);
      }
    }
    __syncthreads();
  }

  // ---- epilogue: C layout col = lane&15, row = r + 8*(lane>>4) ----
  const int c    = lane & 15;
  const int half = lane >> 4;
#pragma unroll
  for (int mi = 0; mi < 2; ++mi)
#pragma unroll
    for (int ni = 0; ni < 4; ++ni)
#pragma unroll
      for (int r = 0; r < 8; ++r) {
        int row = m0 + wr * 32 + mi * 16 + r + 8 * half;
        int col = n0 + wc * 64 + ni * 16 + c;
        if (OUT_BF16)
          ((__bf16*)Cout)[(size_t)row * N + col] = (__bf16)acc[mi][ni][r];
        else
          ((float*)Cout)[(size_t)row * N + col] = acc[mi][ni][r];
      }
}

// ---------------------------------------------------------------------------
// Flash attention: grid (T/256, B*H), block 256 (8 waves).
// Wave owns 32 query rows (2 M-frags); Q frags in registers.
// K tile (64 keys x 64 dims, bf16) staged by TDM with LDS padding that
// reproduces the LDK=72 stride; V^T staged manually (TDM can't transpose).
// qkv row layout per token: [h][3][d] -> q at h*192, k at +64, v at +128.
// ---------------------------------------------------------------------------
__global__ __launch_bounds__(256)
void flash_attn_wmma(const __bf16* __restrict__ qkv, float* __restrict__ O) {
  constexpr int LDK = 72;                      // 64 + 8 pad
  constexpr int BN  = 64;                      // keys per tile
  __shared__ __attribute__((aligned(16))) __bf16 Kls[BN * LDK];       // K[j][d]  == Bt for QK^T
  __shared__ __attribute__((aligned(16))) __bf16 Vtls[D * LDK];       // V^T[d][j] == Bt for PV
  __shared__ __attribute__((aligned(16))) __bf16 Pls[8 * 32 * LDK];   // per-wave P staging

  const int tid  = threadIdx.x;
  const int lane = tid & 31;
  const int wave = tid >> 5;
  const int half = lane >> 4;
  const int b    = blockIdx.y >> 4;
  const int h    = blockIdx.y & 15;
  const int rowQ0 = b * T + blockIdx.x * 256 + wave * 32;   // wave's M=0 row

  // ---- Q fragments straight from global (already bf16) ----
  const __bf16* qbase = qkv + (size_t)rowQ0 * T3 + h * (3 * D);
  v16bf qf[2][2];
#pragma unroll
  for (int mf = 0; mf < 2; ++mf)
#pragma unroll
    for (int kt = 0; kt < 2; ++kt)
      qf[mf][kt] = load_a_frag(qbase + (size_t)mf * 16 * T3, T3, lane, kt * 32);

  float m[2][8], l[2][8];
  v8f acc_o[2][4];
#pragma unroll
  for (int mf = 0; mf < 2; ++mf) {
#pragma unroll
    for (int r = 0; r < 8; ++r) { m[mf][r] = -3.0e38f; l[mf][r] = 0.0f; }
#pragma unroll
    for (int ni = 0; ni < 4; ++ni)
#pragma unroll
      for (int r = 0; r < 8; ++r) acc_o[mf][ni][r] = 0.0f;
  }

  const float scale = 0.125f;                  // 1/sqrt(64)
  const int jl = tid >> 2;                     // 0..63  key row this thread stages
  const int cg = (tid & 3) * 16;               // 0/16/32/48 d-column group

  for (int jt = 0; jt < T / BN; ++jt) {
    const int rowKV = b * T + jt * BN;
    __syncthreads();                           // prior iter done with LDS

    // ---- K tile via Tensor Data Mover (one wave issues the DMA) ----
    if (wave == 0) {
      const unsigned long long gaddr =
          (unsigned long long)(uintptr_t)(qkv + (size_t)rowKV * T3 + h * (3 * D) + D);
      const unsigned int laddr = (unsigned int)(uintptr_t)(&Kls[0]); // LDS offset = addr[31:0]
      u32x4 g0;
      g0[0] = 1u;                                                   // count=1 valid descriptor
      g0[1] = laddr;                                                // lds_addr
      g0[2] = (unsigned int)gaddr;                                  // global_addr[31:0]
      g0[3] = ((unsigned int)(gaddr >> 32) & 0x01FFFFFFu) | (2u << 30); // addr[56:32] | type=2
      i32x8 g1;
      g1[0] = (int)((1u << 16)      // data_size = 2 bytes
                  | (1u << 20)      // pad_enable
                  | (4u << 22)      // pad_interval: 32 DW = 128 B (one 64-elem row)
                  | (3u << 25));    // pad_amount:   4 DW = 16 B  -> LDS stride 144 B (LDK=72)
      g1[1] = (int)(64u << 16);     // tensor_dim0 = 64   (bits 79:48)
      g1[2] = (int)(64u << 16);     // tensor_dim1 = 64   (bits 111:80)
      g1[3] = (int)(64u << 16);     // tile_dim0   = 64   (bits 127:112)
      g1[4] = 64;                   // tile_dim1   = 64   (bits 143:128)
      g1[5] = T3;                   // tensor_dim0_stride = 3072 (bits 207:160)
      g1[6] = 0;
      g1[7] = 0;
      i32x4 gz;
      gz[0] = 0; gz[1] = 0; gz[2] = 0; gz[3] = 0;                   // 2-D tile: groups 2/3 unused
      i32x8 gz8;
#pragma unroll
      for (int i = 0; i < 8; ++i) gz8[i] = 0;
      // amdgpu-toolchain (clang-23) 6-arg form:
      // (u32x4 g0, i32x8 g1, i32x4 g2, i32x4 g3, i32x8 ext, i32 cpol)
      __builtin_amdgcn_tensor_load_to_lds(g0, g1, gz, gz, gz8, 0);
    }

    // ---- V^T staged manually, overlapped with the TDM transfer ----
    {
      const __bf16* vp = qkv + (size_t)(rowKV + jl) * T3 + h * (3 * D) + 2 * D + cg;
      v8bf v0 = *(const v8bf*)vp;
      v8bf v1 = *(const v8bf*)(vp + 8);
#pragma unroll
      for (int i = 0; i < 8; ++i) {
        Vtls[(cg + i) * LDK + jl]     = v0[i];
        Vtls[(cg + 8 + i) * LDK + jl] = v1[i];
      }
    }
    if (wave == 0) __builtin_amdgcn_s_wait_tensorcnt(0);  // K tile landed
    __syncthreads();                                      // K/V visible to all waves

    // ---- S = Q * K^T  (32 x 64): B-frags shared by both M-frags ----
    v8f s[2][4];
#pragma unroll
    for (int mf = 0; mf < 2; ++mf)
#pragma unroll
      for (int ni = 0; ni < 4; ++ni)
#pragma unroll
        for (int r = 0; r < 8; ++r) s[mf][ni][r] = 0.0f;
#pragma unroll
    for (int kt = 0; kt < 2; ++kt)
#pragma unroll
      for (int ni = 0; ni < 4; ++ni) {
        v16bf bf = load_b_frag(&Kls[(ni * 16) * LDK], LDK, lane, kt * 32);
#pragma unroll
        for (int mf = 0; mf < 2; ++mf)
          s[mf][ni] = wmma_bf16(qf[mf][kt], bf, s[mf][ni]);
      }

    // ---- online softmax (rows r+8*half, values across 16 lanes of half) ----
#pragma unroll
    for (int mf = 0; mf < 2; ++mf)
#pragma unroll
      for (int r = 0; r < 8; ++r) {
        float mx = fmaxf(fmaxf(s[mf][0][r], s[mf][1][r]),
                         fmaxf(s[mf][2][r], s[mf][3][r]));
        mx = fmaxf(mx, swz_xor<1>(mx));
        mx = fmaxf(mx, swz_xor<2>(mx));
        mx = fmaxf(mx, swz_xor<4>(mx));
        mx = fmaxf(mx, swz_xor<8>(mx));
        float mnew = fmaxf(m[mf][r], mx * scale);
        float corr = __expf(m[mf][r] - mnew);
        float rsum = 0.0f;
#pragma unroll
        for (int ni = 0; ni < 4; ++ni) {
          float p = __expf(s[mf][ni][r] * scale - mnew);
          s[mf][ni][r] = p;
          rsum += p;
        }
        rsum += swz_xor<1>(rsum);
        rsum += swz_xor<2>(rsum);
        rsum += swz_xor<4>(rsum);
        rsum += swz_xor<8>(rsum);
        l[mf][r] = l[mf][r] * corr + rsum;
        m[mf][r] = mnew;
#pragma unroll
        for (int ni = 0; ni < 4; ++ni) acc_o[mf][ni][r] *= corr;
      }

    // ---- P: C-layout -> A-layout via per-wave LDS staging ----
    __bf16* pw = &Pls[wave * 32 * LDK];
#pragma unroll
    for (int mf = 0; mf < 2; ++mf)
#pragma unroll
      for (int ni = 0; ni < 4; ++ni)
#pragma unroll
        for (int r = 0; r < 8; ++r)
          pw[(mf * 16 + r + 8 * half) * LDK + ni * 16 + (lane & 15)] =
              (__bf16)s[mf][ni][r];
    __syncthreads();                           // P visible (+ K reads done)

    // ---- O += P * V  (P 32x64, V^T staged as Bt; B-frags shared) ----
    v16bf pa[2][2];
#pragma unroll
    for (int mf = 0; mf < 2; ++mf)
#pragma unroll
      for (int kt = 0; kt < 2; ++kt)
        pa[mf][kt] = load_a_frag(pw + mf * 16 * LDK, LDK, lane, kt * 32);
#pragma unroll
    for (int kt = 0; kt < 2; ++kt)
#pragma unroll
      for (int ni = 0; ni < 4; ++ni) {
        v16bf bf = load_b_frag(&Vtls[(ni * 16) * LDK], LDK, lane, kt * 32);
#pragma unroll
        for (int mf = 0; mf < 2; ++mf)
          acc_o[mf][ni] = wmma_bf16(pa[mf][kt], bf, acc_o[mf][ni]);
      }
  }

  // ---- epilogue: O[b*T+t][h*64 + dcol], fp32 for the output GEMM ----
#pragma unroll
  for (int mf = 0; mf < 2; ++mf) {
    float linv[8];
#pragma unroll
    for (int r = 0; r < 8; ++r) linv[r] = 1.0f / l[mf][r];
#pragma unroll
    for (int ni = 0; ni < 4; ++ni)
#pragma unroll
      for (int r = 0; r < 8; ++r) {
        int row = rowQ0 + mf * 16 + r + 8 * half;
        int col = h * D + ni * 16 + (lane & 15);
        O[(size_t)row * E + col] = acc_o[mf][ni][r] * linv[r];
      }
  }
}

// ---------------------------------------------------------------------------
extern "C" void kernel_launch(void* const* d_in, const int* in_sizes, int n_in,
                              void* d_out, int out_size, void* d_ws, size_t ws_size,
                              hipStream_t stream) {
  (void)in_sizes; (void)n_in; (void)out_size; (void)ws_size;
  const float* x     = (const float*)d_in[0];
  const float* W_qkv = (const float*)d_in[1];
  const float* W_out = (const float*)d_in[2];
  float* out = (float*)d_out;

  __bf16* qkv = (__bf16*)d_ws;                                     // 8192 x 3072 bf16 (48 MB)
  float*  Oat = (float*)((char*)d_ws + (size_t)MROWS * T3 * 2);    // 8192 x 1024 f32  (32 MB)

  // 1) qkv = x @ W_qkv  -> bf16
  gemm_wmma_bf16<true><<<dim3(MROWS / 128, T3 / 128), 256, 0, stream>>>(
      x, W_qkv, (void*)qkv, MROWS, T3, E);

  // 2) flash attention over (b,h), O in fp32
  flash_attn_wmma<<<dim3(T / 256, Bsz * H), 256, 0, stream>>>(qkv, Oat);

  // 3) out = O @ W_out -> fp32
  gemm_wmma_bf16<false><<<dim3(MROWS / 128, E / 128), 256, 0, stream>>>(
      Oat, W_out, (void*)out, MROWS, E, E);
}
